// FourierLoss_32220844655148
// MI455X (gfx1250) — matrix operand order repaired and checked by reading, not compiled, not run on hardware
//
#include <hip/hip_runtime.h>
#include <math.h>

// ---------------------------------------------------------------------------
// Fourier contour loss on gfx1250 (CDNA5, wave32, WMMA).
// Pipeline:
//   k_prep      : binarize+pad masks (4 images: pred b0/b1, tgt b0/b1),
//                 init CC labels (flat index), reset map/counters.
//   k_cc  (x63) : LDS-tiled 8-step Jacobi max-propagation => >=504 steps,
//                 same fixed point as the reference's 500 iterations.
//   k_bound     : Laplacian boundary test, labeled contour image, label
//                 registration (atomicCAS dense map) + contour pixel counts.
//   k_sortc     : sort component labels ascending (np.unique order).
//   k_comp      : per-component: ordered compaction, centroid, bitonic sort
//                 by angle, delta/arc-length scan, 8-harmonic Fourier sums
//                 via v_wmma_f32_16x16x32_f16 (A=trig 16x32, B=delta bcast).
//                 Operand build is branch-free: zero-padded LDS + sin phase
//                 shift (cos(x) = sin(x + pi/2)).
//   k_loss      : nearest-center matching + scaled L1 amplitude loss.
// ---------------------------------------------------------------------------

typedef __attribute__((ext_vector_type(16))) _Float16 v16h;
typedef __attribute__((ext_vector_type(8)))  float    v8f;

#define PH   514
#define PPX  (514 * 514)      // 264196 padded pixels per image
#define CW   512
#define CPX  (512 * 512)      // 262144 cropped pixels per image
#define NIMG 4
#define MAXC 96
#define MAXP 2048
#define TILE 32
#define HALO 8
#define LW   48               // TILE + 2*HALO

#define PI_F     3.14159265358979323846f
#define HPI_F    1.57079632679489661923f
#define TWOPI_F  6.28318530717958647692f

// --------------------------------------------------------------------------
__global__ void k_prep(const float* __restrict__ pred,
                       const float* __restrict__ tgt,
                       float* __restrict__ fg, float* __restrict__ ccA,
                       int* __restrict__ map, int* __restrict__ cnts,
                       int* __restrict__ labA) {
  int i = blockIdx.x * 256 + threadIdx.x;
  if (i < NIMG) {              // counters: compCount=1 (slot0 = label 0)
    cnts[i] = 1;
    cnts[NIMG + i] = 0;        // contour pixel count
    labA[i * MAXC] = 0;        // label-0 component always present
  }
  if (i >= NIMG * PPX) return;
  map[i] = -1;
  int s = i / PPX, r = i % PPX;
  int py = r / PH, px = r % PH;
  float f = 0.0f;
  if (py >= 1 && py < PH - 1 && px >= 1 && px < PH - 1) {
    int b = s & 1;
    int src = b * CPX + (py - 1) * CW + (px - 1);
    if (s < 2) f = (pred[src] > 0.0f) ? 1.0f : 0.0f;   // sigmoid(x)>0.5 <=> x>0
    else       f = (tgt[src] == 1.0f) ? 1.0f : 0.0f;
  }
  fg[i] = f;
  ccA[i] = (f != 0.0f) ? (float)((s & 1) * PPX + r) : 0.0f;  // arange init
}

// --------------------------------------------------------------------------
// 8 Jacobi steps of fg-masked 3x3 max propagation per launch (double buffered
// in LDS; interior TILE region written back, halo guarantees validity).
__global__ __launch_bounds__(256) void k_cc(const float* __restrict__ src,
                                            float* __restrict__ dst,
                                            const float* __restrict__ fg) {
  __shared__ float sF[LW * LW], s0[LW * LW], s1[LW * LW];
  int s  = blockIdx.z;
  int bx = blockIdx.x * TILE - HALO, by = blockIdx.y * TILE - HALO;
  int tid = threadIdx.x;
  const float* F = fg  + s * PPX;
  const float* S = src + s * PPX;
  for (int c = tid; c < LW * LW; c += 256) {
    int ly = c / LW, lx = c % LW;
    int gy = by + ly, gx = bx + lx;
    bool in = (gx >= 0 && gx < PH && gy >= 0 && gy < PH);
    sF[c] = in ? F[gy * PH + gx] : 0.0f;
    s0[c] = in ? S[gy * PH + gx] : 0.0f;
  }
  __syncthreads();
  float* a = s0; float* b = s1;
  for (int it = 0; it < HALO; ++it) {
    for (int c = tid; c < LW * LW; c += 256) {
      int ly = c / LW, lx = c % LW;
      float v = a[c];
      if (lx >= 1 && lx < LW - 1 && ly >= 1 && ly < LW - 1 && sF[c] != 0.0f) {
        float m = v;
#pragma unroll
        for (int dy = -1; dy <= 1; ++dy)
#pragma unroll
          for (int dx = -1; dx <= 1; ++dx)
            m = fmaxf(m, a[(ly + dy) * LW + lx + dx]);
        v = m;
      }
      b[c] = v;
    }
    __syncthreads();
    float* t = a; a = b; b = t;
  }
  float* D = dst + s * PPX;
  for (int c = tid; c < LW * LW; c += 256) {
    int ly = c / LW, lx = c % LW;
    if (lx >= HALO && lx < HALO + TILE && ly >= HALO && ly < HALO + TILE) {
      int gy = by + ly, gx = bx + lx;
      if (gy < PH && gx < PH) D[gy * PH + gx] = a[c];
    }
  }
}

// --------------------------------------------------------------------------
__global__ void k_bound(const float* __restrict__ fg,
                        const float* __restrict__ cc,
                        float* __restrict__ LC, int* __restrict__ map,
                        int* __restrict__ cnts, int* __restrict__ labA) {
  int i = blockIdx.x * 256 + threadIdx.x;
  if (i >= NIMG * CPX) return;
  int s = i >> 18;                 // CPX == 2^18
  int r = i & (CPX - 1);
  int y = r >> 9, x = r & 511;
  const float* F = fg + s * PPX;
  int p = (y + 1) * PH + (x + 1);
  float c = F[p];
  float sum = 0.0f;
#pragma unroll
  for (int dy = -1; dy <= 1; ++dy)
#pragma unroll
    for (int dx = -1; dx <= 1; ++dx) sum += F[p + dy * PH + dx];
  float conv = sum - 9.0f * c;     // == (neighbors) - 8*center
  float lab = (conv < 0.0f) ? cc[s * PPX + p] : 0.0f;
  LC[i] = lab;
  if (lab != 0.0f) {
    atomicAdd(&cnts[NIMG + s], 1);
    int gi = (int)lab;
    int li = gi - (s & 1) * PPX;
    int old = atomicCAS(&map[s * PPX + li], -1, -2);
    if (old == -1) {
      int slot = atomicAdd(&cnts[s], 1);
      if (slot < MAXC) labA[s * MAXC + slot] = gi;
    }
  }
}

// --------------------------------------------------------------------------
__global__ void k_sortc(int* __restrict__ labA, const int* __restrict__ cnts) {
  int s = blockIdx.x;
  if (threadIdx.x != 0) return;
  int n = cnts[s]; if (n > MAXC) n = MAXC;
  int* L = labA + s * MAXC;
  for (int i = 1; i < n; ++i) {            // insertion sort (<=96 labels)
    int key = L[i]; int j = i - 1;
    while (j >= 0 && L[j] > key) { L[j + 1] = L[j]; --j; }
    L[j + 1] = key;
  }
}

// --------------------------------------------------------------------------
// One workgroup per component.
__global__ __launch_bounds__(256) void k_comp(const float* __restrict__ LC,
                                              const int* __restrict__ labA,
                                              const int* __restrict__ cnts,
                                              float* __restrict__ recs) {
  int s = blockIdx.y;
  int ci = blockIdx.x;
  int ncomp = cnts[s]; if (ncomp > MAXC) ncomp = MAXC;
  if (ci >= ncomp) return;
  int tid = threadIdx.x;
  float labF = (float)labA[s * MAXC + ci];
  const float* img = LC + s * CPX;

  __shared__ int   sIdx[MAXP];
  __shared__ float sA[MAXP], sB[MAXP], sC[MAXP];
  __shared__ float red[256];
  __shared__ int   wcnt[8];
  __shared__ int   baseCnt;
  __shared__ float shCY, shCX, shL;
  __shared__ int   shN;

  if (tid == 0) baseCnt = 0;
  __syncthreads();

  // --- ordered (row-major) stream compaction of matching pixels -----------
  int wid = tid >> 5, lane = tid & 31;
  int   myCnt = 0; float mySY = 0.f, mySX = 0.f;
  for (int ch = 0; ch < CPX / 256; ++ch) {
    int p = (ch << 8) + tid;
    bool match = (img[p] == labF);
    unsigned mask = (unsigned)__ballot(match);
    if (lane == 0) wcnt[wid] = __popc(mask);
    __syncthreads();
    int off = baseCnt;
    for (int w = 0; w < wid; ++w) off += wcnt[w];
    if (match) {
      int pos = off + __popc(mask & ((1u << lane) - 1u));
      if (pos < MAXP) sIdx[pos] = p;
      myCnt++; mySY += (float)(p >> 9); mySX += (float)(p & 511);
    }
    __syncthreads();
    if (tid == 0) { int t = 0; for (int w = 0; w < 8; ++w) t += wcnt[w]; baseCnt += t; }
    __syncthreads();
  }

  // --- deterministic tree reductions: count, sumY, sumX -------------------
  red[tid] = (float)myCnt; __syncthreads();
  for (int sh = 128; sh > 0; sh >>= 1) { if (tid < sh) red[tid] += red[tid + sh]; __syncthreads(); }
  if (tid == 0) shN = (int)(red[0] + 0.5f);
  __syncthreads();
  red[tid] = mySY; __syncthreads();
  for (int sh = 128; sh > 0; sh >>= 1) { if (tid < sh) red[tid] += red[tid + sh]; __syncthreads(); }
  if (tid == 0) shCY = red[0];
  __syncthreads();
  red[tid] = mySX; __syncthreads();
  for (int sh = 128; sh > 0; sh >>= 1) { if (tid < sh) red[tid] += red[tid + sh]; __syncthreads(); }
  if (tid == 0) shCX = red[0];
  __syncthreads();

  int nAll = shN;
  int rb = (s * MAXC + ci) * 12;
  if (nAll == 0) {
    if (tid == 0) { for (int k = 0; k < 10; ++k) recs[rb + k] = 0.0f; }
    return;
  }
  float cy = shCY / (float)nAll, cx = shCX / (float)nAll;
  int n = (nAll < MAXP) ? nAll : MAXP;
  int n2 = 2; while (n2 < n) n2 <<= 1;

  // --- angles (atan2(dx,dy) as in reference) + sentinel padding -----------
  for (int i = tid; i < n2; i += 256) {
    if (i < n) {
      int id = sIdx[i];
      float dy = (float)(id >> 9) - cy;
      float dx = (float)(id & 511) - cx;
      sA[i] = atan2f(dx, dy);
    } else { sA[i] = 3.0e38f; sIdx[i] = 0x7fffffff; }
  }
  __syncthreads();

  // --- bitonic sort by (angle, row-major index) => stable -----------------
  for (int k = 2; k <= n2; k <<= 1) {
    for (int j = k >> 1; j > 0; j >>= 1) {
      for (int i = tid; i < n2; i += 256) {
        int ixj = i ^ j;
        if (ixj > i) {
          float a0 = sA[i], a1 = sA[ixj];
          int   i0 = sIdx[i], i1 = sIdx[ixj];
          bool gt = (a0 > a1) || (a0 == a1 && i0 > i1);
          bool up = ((i & k) == 0);
          if (gt == up) { sA[i] = a1; sA[ixj] = a0; sIdx[i] = i1; sIdx[ixj] = i0; }
        }
      }
      __syncthreads();
    }
  }

  // --- d1 (radial), delta, segment length d3 ------------------------------
  for (int i = tid; i < n; i += 256) {
    int id = sIdx[i];
    float dy = (float)(id >> 9) - cy;
    float dx = (float)(id & 511) - cx;
    sC[i] = sqrtf(dx * dx + dy * dy);
  }
  __syncthreads();
  for (int i = tid; i < n; i += 256) {
    int nx = (i + 1 == n) ? 0 : i + 1;
    sB[i] = sC[i] - sC[nx];                          // delta
    int a0 = sIdx[i], a1 = sIdx[nx];
    float ddy = (float)(a0 >> 9) - (float)(a1 >> 9);
    float ddx = (float)(a0 & 511) - (float)(a1 & 511);
    sA[i] = sqrtf(ddx * ddx + ddy * ddy);            // d3
  }
  __syncthreads();

  // --- inclusive scan of d3 -> arc length l -------------------------------
  float* srcp = sA; float* dstp = sC;
  for (int off = 1; off < n; off <<= 1) {
    for (int i = tid; i < n; i += 256)
      dstp[i] = srcp[i] + ((i >= off) ? srcp[i - off] : 0.0f);
    __syncthreads();
    float* t = srcp; srcp = dstp; dstp = t;
  }
  if (srcp != sA) {
    for (int i = tid; i < n; i += 256) sA[i] = srcp[i];
    __syncthreads();
  }
  if (tid == 0) { float Lv = sA[n - 1]; shL = (Lv != 0.0f) ? Lv : 1.0f; }
  // zero-pad l (sA) and delta (sB) up to the next multiple of 32 so the
  // WMMA operand build below is completely branch-free (tail K-slots give
  // sin(0)*0 == 0, no NaN risk from stale sort sentinels).
  {
    int npad = (n + 31) & ~31;        // <= MAXP, always in bounds
    for (int i = n + tid; i < npad; i += 256) { sA[i] = 0.0f; sB[i] = 0.0f; }
  }
  __syncthreads();
  float Lval = shL;

  // --- Fourier projection via V_WMMA_F32_16X16X32_F16 (wave 0 only) -------
  // A[m][K]: rows 0..7 = sin((m+1)*w0*l), rows 8..15 = cos => sin(x + pi/2).
  // B[K][N] = delta[K] broadcast over all 16 columns => D columns all equal.
  if (tid < 32) {
    int L = tid;
    int m  = L & 15;
    int hi = L >> 4;
    float kh = (float)((m & 7) + 1);
    float phOff = (m < 8) ? 0.0f : HPI_F;     // cos(x) = sin(x + pi/2)
    float w0 = TWOPI_F / Lval;
    float wk = w0 * kh;
    v8f acc = {};
    int nch = (n + 31) >> 5;
    for (int c = 0; c < nch; ++c) {
      int base = c << 5;
      v16h a; v16h bm;
#pragma unroll
      for (int v = 0; v < 8; ++v) {
        int kb = ((v >> 2) << 4) + (hi << 3) + ((v & 3) << 1);  // A K-pair base
        int Kb = ((v >> 2) << 4) + (hi << 3) + ((v & 3) << 1);  // B K-pair base
#pragma unroll
        for (int e = 0; e < 2; ++e) {
          float ph = fmaf(wk, sA[base + kb + e], phOff);
          a[2 * v + e]  = (_Float16)__sinf(ph);
          bm[2 * v + e] = (_Float16)sB[base + Kb + e];
        }
      }
      acc = __builtin_amdgcn_wmma_f32_16x16x32_f16(
          false, a, false, bm, (short)0, acc, false, false);
    }
    if (L == 0) {
#pragma unroll
      for (int j = 0; j < 8; ++j) sC[j] = acc[j];       // S_sin[1..8]
    } else if (L == 16) {
#pragma unroll
      for (int j = 0; j < 8; ++j) sC[8 + j] = acc[j];   // S_cos[1..8]
    }
  }
  __syncthreads();

  if (tid < 8) {
    float kk = (float)(tid + 1);
    float av =  sC[tid]     / (kk * PI_F);
    float bv = -sC[8 + tid] / (kk * PI_F);
    recs[rb + 2 + tid] = sqrtf(av * av + bv * bv);
  }
  if (tid == 0) { recs[rb + 0] = cy; recs[rb + 1] = cx; }
}

// --------------------------------------------------------------------------
__global__ void k_loss(const float* __restrict__ recs,
                       const int* __restrict__ cnts,
                       const float* __restrict__ w, float* __restrict__ out) {
  if (threadIdx.x != 0 || blockIdx.x != 0) return;
  float wv = w[0];
  float scale = 0.5f * wv * wv;
  float shift = logf(1.0f + wv * wv);
  float total = 0.0f;
  for (int b = 0; b < 2; ++b) {
    int sp = b, st = 2 + b;
    if (cnts[NIMG + sp] == 0 || cnts[NIMG + st] == 0) continue;  // sum==0 skip
    int nP = cnts[sp]; if (nP > MAXC) nP = MAXC;
    int nT = cnts[st]; if (nT > MAXC) nT = MAXC;
    if (nT <= nP) {
      for (int t = 0; t < nT; ++t) {
        float gy = recs[(st * MAXC + t) * 12 + 0];
        float gx = recs[(st * MAXC + t) * 12 + 1];
        int best = 0; float bd = 3.0e38f;
        for (int p = 0; p < nP; ++p) {
          float dy = gy - recs[(sp * MAXC + p) * 12 + 0];
          float dx = gx - recs[(sp * MAXC + p) * 12 + 1];
          float d = dy * dy + dx * dx;
          if (d < bd) { bd = d; best = p; }
        }
        for (int k = 0; k < 8; ++k) {
          float pd = recs[(sp * MAXC + best) * 12 + 2 + k] * scale + shift;
          float gd = recs[(st * MAXC + t)   * 12 + 2 + k] * scale + shift;
          total += fabsf(pd - gd);
        }
      }
    } else {
      for (int p = 0; p < nP; ++p) {
        float py = recs[(sp * MAXC + p) * 12 + 0];
        float px = recs[(sp * MAXC + p) * 12 + 1];
        int best = 0; float bd = 3.0e38f;
        for (int t = 0; t < nT; ++t) {
          float dy = py - recs[(st * MAXC + t) * 12 + 0];
          float dx = px - recs[(st * MAXC + t) * 12 + 1];
          float d = dy * dy + dx * dx;
          if (d < bd) { bd = d; best = t; }
        }
        for (int k = 0; k < 8; ++k) {
          float pd = recs[(sp * MAXC + p)    * 12 + 2 + k] * scale + shift;
          float gd = recs[(st * MAXC + best) * 12 + 2 + k] * scale + shift;
          total += fabsf(pd - gd);
        }
      }
    }
  }
  out[0] = total;
}

// --------------------------------------------------------------------------
extern "C" void kernel_launch(void* const* d_in, const int* in_sizes, int n_in,
                              void* d_out, int out_size, void* d_ws, size_t ws_size,
                              hipStream_t stream) {
  (void)in_sizes; (void)n_in; (void)out_size; (void)ws_size;
  const float* pred = (const float*)d_in[0];
  const float* tgt  = (const float*)d_in[1];
  const float* wgt  = (const float*)d_in[2];
  float* out = (float*)d_out;

  const size_t NP = (size_t)NIMG * PPX;   // 1,056,784
  float* fg   = (float*)d_ws;
  float* ccA  = fg  + NP;
  float* ccB  = ccA + NP;
  float* LC   = ccB + NP;                 // NIMG * CPX floats
  float* recs = LC + (size_t)NIMG * CPX;  // NIMG * MAXC * 12 floats
  int*   map  = (int*)(recs + NIMG * MAXC * 12);
  int*   labA = map + NP;
  int*   cnts = labA + NIMG * MAXC;       // [0..3]=compCount, [4..7]=contourPix

  k_prep<<<(int)((NP + 255) / 256), 256, 0, stream>>>(pred, tgt, fg, ccA, map,
                                                      cnts, labA);
  float* a = ccA; float* b = ccB;
  for (int it = 0; it < 63; ++it) {       // 63 * 8 = 504 >= 500 steps
    k_cc<<<dim3(17, 17, NIMG), 256, 0, stream>>>(a, b, fg);
    float* t = a; a = b; b = t;
  }
  k_bound<<<(NIMG * CPX + 255) / 256, 256, 0, stream>>>(fg, a, LC, map, cnts,
                                                        labA);
  k_sortc<<<NIMG, 32, 0, stream>>>(labA, cnts);
  k_comp<<<dim3(MAXC, NIMG), 256, 0, stream>>>(LC, labA, cnts, recs);
  k_loss<<<1, 32, 0, stream>>>(recs, cnts, wgt, out);
}
